// TFMamba_74526272520202
// MI455X (gfx1250) — matrix operand order repaired and checked by reading, not compiled, not run on hardware
//
#include <hip/hip_runtime.h>
#include <hip/hip_bf16.h>
#include <math.h>

// ---------------- problem constants (from reference) ----------------
#define BB        2
#define LL        1024
#define D_MODEL   1024
#define D_INNER   2048
#define D_STATE   16
#define DT_RANK   64
#define D_CONV    4
#define NTOK      (BB * LL)            // 2048 rows for all GEMMs

// ---------------- CDNA5 WMMA types ----------------
typedef __attribute__((ext_vector_type(16))) __bf16 v16bf;
typedef __attribute__((ext_vector_type(8)))  __bf16 v8bf;
typedef __attribute__((ext_vector_type(8)))  float  v8f;

// gfx1250 async global->LDS (ASYNCcnt) support, feature-gated
#if __has_builtin(__builtin_amdgcn_global_load_async_to_lds_b32) && \
    __has_builtin(__builtin_amdgcn_s_wait_asynccnt)
#define HAS_ASYNC_LDS 1
typedef __attribute__((address_space(1))) int* gas_i32_t;   // global int*
typedef __attribute__((address_space(3))) int* las_i32_t;   // LDS int*
#else
#define HAS_ASYNC_LDS 0
#endif

// pack two fp32 into two bf16 (one u32), round-to-nearest
__device__ __forceinline__ unsigned int pack2bf(float a, float b) {
#if __has_builtin(__builtin_amdgcn_cvt_pk_bf16_f32)
    auto p = __builtin_amdgcn_cvt_pk_bf16_f32(a, b);   // v_cvt_pk_bf16_f32
    return __builtin_bit_cast(unsigned int, p);
#else
    // 2 adds + 1 v_perm_b32: round-to-nearest then merge high halves
    unsigned int ua = __builtin_bit_cast(unsigned int, a) + 0x8000u;
    unsigned int ub = __builtin_bit_cast(unsigned int, b) + 0x8000u;
    return __builtin_amdgcn_perm(ub, ua, 0x07060302u); // {ub.hi, ua.hi}
#endif
}

__device__ __forceinline__ float siluf(float x) {
    return x / (1.0f + __expf(-x));
}
__device__ __forceinline__ float softplusf(float x) {
    return (x > 20.0f) ? x : log1pf(__expf(x));
}

// =====================================================================
// Block-tiled WMMA GEMM: C[M,N] = epi( A[M,K](f32) * W[K,N](f32) )
// Block = 256 thr (8 waves). Block tile 128(M) x 64(N), K-step 32.
// fp32 -> bf16 conversion happens ONCE per block into LDS:
//   sA[m][k]   row-major   (128x32 bf16, 8KB)
//   sB[n][k]   TRANSPOSED  ( 64x32 bf16, 4KB)
// Each wave computes a 32x32 C macro-tile = 2x2 v_wmma_f32_16x16x32_bf16
// per K-step; fragments come from contiguous ds_load_b128 pairs.
// mode 0: plain store;  mode 1: softplus(c + bias[n])
// M must be a multiple of 128, K a multiple of 32; N is guarded.
// =====================================================================
__global__ __launch_bounds__(256)
void wmma_gemm_bf16(const float* __restrict__ A, int lda,
                    const float* __restrict__ W, int ldb,
                    float* __restrict__ C, int ldc,
                    int M, int N, int K,
                    int mode, const float* __restrict__ bias)
{
    __shared__ __align__(16) unsigned short sA[128 * 32];
    __shared__ __align__(16) unsigned short sB[64 * 32];

    const int tid  = threadIdx.x;
    const int lane = tid & 31;
    const int wave = tid >> 5;
    const int wm   = wave & 3;      // 4 waves along M (32 rows each)
    const int wn   = wave >> 2;     // 2 waves along N (32 cols each)
    const int m0   = blockIdx.y * 128;
    const int n0   = blockIdx.x * 64;

    const int kh = lane >> 4;       // half-wave selector
    const int ln = lane & 15;       // row/col within 16x16 tile

    v8f acc[2][2] = {};

    for (int k0 = 0; k0 < K; k0 += 32) {
        // ---- stage A tile: 128x32 f32 -> bf16 LDS (8 u32/thread) ----
        {
            if (k0 + 32 < K)
                __builtin_prefetch(A + (size_t)(m0 + (tid >> 1)) * lda + k0 + 32, 0, 0);
            unsigned int* sA32 = (unsigned int*)sA;
            #pragma unroll
            for (int j = 0; j < 8; ++j) {
                int p  = tid + 256 * j;      // 0..2047 bf16-pairs
                int m  = p >> 4;             // 0..127
                int kp = p & 15;             // pair index, k = 2*kp
                const float2 v = ((const float2*)(A + (size_t)(m0 + m) * lda + k0))[kp];
                sA32[m * 16 + kp] = pack2bf(v.x, v.y);
            }
        }
        // ---- stage B tile transposed: 32x64 f32 -> sB[n][k] (4 u32/thread) ----
        {
            unsigned int* sB32 = (unsigned int*)sB;
            #pragma unroll
            for (int j = 0; j < 4; ++j) {
                int q  = tid + 256 * j;      // 0..1023
                int kp = q >> 6;             // 0..15 (k = 2*kp)
                int n  = q & 63;
                int nc = n0 + n; if (nc > N - 1) nc = N - 1;   // clamp (ragged N)
                float w0 = W[(size_t)(k0 + 2 * kp)     * ldb + nc];
                float w1 = W[(size_t)(k0 + 2 * kp + 1) * ldb + nc];
                sB32[n * 16 + kp] = pack2bf(w0, w1);
            }
        }
        __syncthreads();

        // ---- fragments from LDS (ds_load_b128 pairs) ----
        v16bf afrag[2], bfrag[2];
        #pragma unroll
        for (int mi = 0; mi < 2; ++mi) {
            int r = wm * 32 + mi * 16 + ln;                 // A row in block tile
            v8bf lo = *(const v8bf*)&sA[r * 32 + 8 * kh];        // k = 8kh..8kh+7
            v8bf hi = *(const v8bf*)&sA[r * 32 + 16 + 8 * kh];   // k = 16+8kh..
            afrag[mi] = __builtin_shufflevector(lo, hi,
                0, 1, 2, 3, 4, 5, 6, 7, 8, 9, 10, 11, 12, 13, 14, 15);
        }
        #pragma unroll
        for (int ni = 0; ni < 2; ++ni) {
            int cn = wn * 32 + ni * 16 + ln;                // B col in block tile
            v8bf lo = *(const v8bf*)&sB[cn * 32 + 8 * kh];       // k = 8kh..8kh+7
            v8bf hi = *(const v8bf*)&sB[cn * 32 + 16 + 8 * kh];
            bfrag[ni] = __builtin_shufflevector(lo, hi,
                0, 1, 2, 3, 4, 5, 6, 7, 8, 9, 10, 11, 12, 13, 14, 15);
        }

        // ---- 2x2 WMMA ----
        #pragma unroll
        for (int mi = 0; mi < 2; ++mi)
            #pragma unroll
            for (int ni = 0; ni < 2; ++ni)
                acc[mi][ni] = __builtin_amdgcn_wmma_f32_16x16x32_bf16(
                    false, afrag[mi], false, bfrag[ni],
                    (short)0, acc[mi][ni], false, false);

        __syncthreads();
    }

    // ---- store (C layout: VGPR r -> row r + 8*kh, col ln) ----
    #pragma unroll
    for (int ni = 0; ni < 2; ++ni) {
        int col = n0 + wn * 32 + ni * 16 + ln;
        if (col < N) {
            float bv = (mode == 1) ? bias[col] : 0.0f;
            #pragma unroll
            for (int mi = 0; mi < 2; ++mi) {
                int mbase = m0 + wm * 32 + mi * 16 + 8 * kh;
                #pragma unroll
                for (int rr = 0; rr < 8; ++rr) {
                    float cv = acc[mi][ni][rr];
                    if (mode == 1) cv = softplusf(cv + bv);
                    C[(size_t)(mbase + rr) * ldc + col] = cv;
                }
            }
        }
    }
}

// =====================================================================
// Causal depthwise conv (k=4) + SiLU over x = xz[:, :D_INNER]
// =====================================================================
__global__ __launch_bounds__(256)
void conv_silu_kernel(const float* __restrict__ xz,      // (NTOK, 2*D_INNER)
                      const float* __restrict__ kern,    // (D_CONV, D_INNER)
                      const float* __restrict__ bias,    // (D_INNER)
                      float* __restrict__ u)             // (NTOK, D_INNER)
{
    int idx = blockIdx.x * blockDim.x + threadIdx.x;
    if (idx >= NTOK * D_INNER) return;
    int d = idx & (D_INNER - 1);
    int t = idx / D_INNER;          // b*L + l
    int l = t & (LL - 1);
    int b = t >> 10;                // L = 1024

    float acc = bias[d];
    #pragma unroll
    for (int i = 0; i < D_CONV; ++i) {
        int ls = l - (D_CONV - 1) + i;
        if (ls >= 0)
            acc += xz[((size_t)(b * LL + ls)) * (2 * D_INNER) + d] *
                   kern[i * D_INNER + d];
    }
    u[idx] = siluf(acc);
}

// =====================================================================
// Selective scan: sequential over L, parallel over (b, d).
// 16 states per channel live in registers. Per-step B/C vectors
// (shared by all 2048 channels of a batch) are DOUBLE-BUFFERED in LDS
// and prefetched with GLOBAL_LOAD_ASYNC_TO_LDS_B32 (ASYNCcnt) one
// timestep ahead of the compute.
// Fused epilogue: y = (scan + u*Dskip) * silu(z)
// =====================================================================
__global__ __launch_bounds__(256)
void selective_scan_kernel(const float* __restrict__ u,      // (NTOK, D_INNER)
                           const float* __restrict__ delta,  // (NTOK, D_INNER)
                           const float* __restrict__ xdbl,   // (NTOK, 96)
                           const float* __restrict__ xz,     // (NTOK, 2*D_INNER)
                           const float* __restrict__ A_log,  // (D_INNER, D_STATE)
                           const float* __restrict__ Dskip,  // (D_INNER)
                           float* __restrict__ y_act)        // (NTOK, D_INNER)
{
    const int b   = blockIdx.y;
    const int tid = threadIdx.x;
    const int d   = blockIdx.x * blockDim.x + tid;

    float Arow[D_STATE];
    #pragma unroll
    for (int n = 0; n < D_STATE; ++n)
        Arow[n] = -__expf(A_log[(size_t)d * D_STATE + n]);

    float st[D_STATE];
    #pragma unroll
    for (int n = 0; n < D_STATE; ++n) st[n] = 0.0f;

    const float Dv = Dskip[d];

    __shared__ float sBC[2][2 * D_STATE];   // double buffer: [0..15]=B, [16..31]=C

    // issue fetch of timestep l's B/C into buffer `buf` (lanes 0..31 of wave 0)
    auto issue_bc = [&](int l, int buf) {
        if (tid < 2 * D_STATE) {
            const float* src = xdbl + ((size_t)b * LL + l) * (DT_RANK + 2 * D_STATE)
                                    + DT_RANK + tid;
#if HAS_ASYNC_LDS
            __builtin_amdgcn_global_load_async_to_lds_b32(
                (gas_i32_t)(void*)(const_cast<float*>(src)),
                (las_i32_t)(void*)&sBC[buf][tid], 0, 0);
#else
            sBC[buf][tid] = *src;
#endif
        }
    };

    issue_bc(0, 0);
#if HAS_ASYNC_LDS
    if (tid < 2 * D_STATE) __builtin_amdgcn_s_wait_asynccnt(0);
#endif
    __syncthreads();

    int buf = 0;
    for (int l = 0; l < LL; ++l) {
        // prefetch next step's B/C into the other buffer while we compute
        if (l + 1 < LL) issue_bc(l + 1, buf ^ 1);

        const size_t row = (size_t)b * LL + l;
        const float dl = delta[row * D_INNER + d];
        const float uu = u[row * D_INNER + d];
        const float* bc = sBC[buf];

        float acc = 0.0f;
        #pragma unroll
        for (int n = 0; n < D_STATE; ++n) {
            float dA = __expf(dl * Arow[n]);
            st[n] = dA * st[n] + (dl * bc[n]) * uu;
            acc += st[n] * bc[D_STATE + n];
        }

        const float zz = xz[row * (2 * D_INNER) + D_INNER + d];
        y_act[row * D_INNER + d] = (acc + uu * Dv) * siluf(zz);

#if HAS_ASYNC_LDS
        if (tid < 2 * D_STATE) __builtin_amdgcn_s_wait_asynccnt(0);
#endif
        __syncthreads();
        buf ^= 1;
    }
}

// =====================================================================
// launcher
// =====================================================================
extern "C" void kernel_launch(void* const* d_in, const int* in_sizes, int n_in,
                              void* d_out, int out_size, void* d_ws, size_t ws_size,
                              hipStream_t stream)
{
    const float* hs      = (const float*)d_in[0];  // (B,L,D_MODEL)
    const float* Win     = (const float*)d_in[1];  // (D_MODEL, 2*D_INNER)
    const float* Wx      = (const float*)d_in[2];  // (D_INNER, 96)
    const float* Wdt     = (const float*)d_in[3];  // (DT_RANK, D_INNER)
    const float* dt_bias = (const float*)d_in[4];  // (D_INNER)
    const float* Wout    = (const float*)d_in[5];  // (D_INNER, D_MODEL)
    const float* dwk     = (const float*)d_in[6];  // (D_CONV, D_INNER)
    const float* dwb     = (const float*)d_in[7];  // (D_INNER)
    const float* A_log   = (const float*)d_in[8];  // (D_INNER, D_STATE)
    const float* Dskip   = (const float*)d_in[9];  // (D_INNER)
    float* out = (float*)d_out;

    // workspace layout (floats)
    float* ws    = (float*)d_ws;
    float* xz    = ws;                                   // NTOK * 2*D_INNER
    float* u     = xz    + (size_t)NTOK * 2 * D_INNER;   // NTOK * D_INNER
    float* xdbl  = u     + (size_t)NTOK * D_INNER;       // NTOK * 96
    float* delta = xdbl  + (size_t)NTOK * (DT_RANK + 2 * D_STATE);
    float* y_act = delta + (size_t)NTOK * D_INNER;       // NTOK * D_INNER

    dim3 blk(256);

    // 1) xz = hs @ Win   : M=2048 K=1024 N=4096
    wmma_gemm_bf16<<<dim3((2 * D_INNER) / 64, NTOK / 128), blk, 0, stream>>>(
        hs, D_MODEL, Win, 2 * D_INNER, xz, 2 * D_INNER,
        NTOK, 2 * D_INNER, D_MODEL, 0, nullptr);

    // 2) u = silu(causal_dwconv(x))
    {
        int n = NTOK * D_INNER;
        conv_silu_kernel<<<dim3((n + 255) / 256), blk, 0, stream>>>(
            xz, dwk, dwb, u);
    }

    // 3) x_dbl = u @ Wx  : M=2048 K=2048 N=96 (ragged N, guarded)
    wmma_gemm_bf16<<<dim3(2, NTOK / 128), blk, 0, stream>>>(
        u, D_INNER, Wx, DT_RANK + 2 * D_STATE,
        xdbl, DT_RANK + 2 * D_STATE,
        NTOK, DT_RANK + 2 * D_STATE, D_INNER, 0, nullptr);

    // 4) delta = softplus(dt @ Wdt + dt_bias) : M=2048 K=64 N=2048
    wmma_gemm_bf16<<<dim3(D_INNER / 64, NTOK / 128), blk, 0, stream>>>(
        xdbl, DT_RANK + 2 * D_STATE, Wdt, D_INNER, delta, D_INNER,
        NTOK, D_INNER, DT_RANK, 1, dt_bias);

    // 5) selective scan + skip + SiLU gate
    selective_scan_kernel<<<dim3(D_INNER / 256, BB), blk, 0, stream>>>(
        u, delta, xdbl, xz, A_log, Dskip, y_act);

    // 6) out = y_act @ Wout : M=2048 K=2048 N=1024
    wmma_gemm_bf16<<<dim3(D_MODEL / 64, NTOK / 128), blk, 0, stream>>>(
        y_act, D_INNER, Wout, D_MODEL, out, D_MODEL,
        NTOK, D_MODEL, D_INNER, 0, nullptr);
}
